// ODERNN_66967130079755
// MI455X (gfx1250) — matrix operand order repaired and checked by reading, not compiled, not run on hardware
//
#include <hip/hip_runtime.h>

// ---------------------------------------------------------------------------
// ODE-RNN scan for MI455X (gfx1250, wave32, WMMA).
// One wave owns a 16-row batch tile across all 64 timesteps; h lives in
// registers as 8x v8f WMMA C-fragments; activations stream through per-wave
// LDS f16 buffers; weights live in LDS (f16) except W_hh (f16 in d_ws / L2).
// Round 2: fuse RK4 k-consumption into layer-2 tile loop to kill the kf[8]
// fragment array (-64 live VGPRs) and the s_set_vgpr_msb thrashing.
// ---------------------------------------------------------------------------

typedef _Float16 half_t;
typedef __attribute__((ext_vector_type(16))) _Float16 v16h;
typedef __attribute__((ext_vector_type(8)))  _Float16 v8h;
typedef __attribute__((ext_vector_type(4)))  _Float16 v4h;
typedef __attribute__((ext_vector_type(8)))  float    v8f;

#define BATCH 1024
#define WLEN  64
#define DXDIM 64
#define HDIM  128
#define CIN   192   // H + DX (reference concatenates [h, ctrl])
#define OHDIM 128
#define G3H   384   // 3*H (gate order r, z, n)

#define WAVES_PER_BLOCK 2
#define THREADS (WAVES_PER_BLOCK * 32)
#define ROWS_PER_BLOCK (WAVES_PER_BLOCK * 16)

__device__ __forceinline__ v8f zero_v8f() {
  v8f c;
#pragma unroll
  for (int j = 0; j < 8; ++j) c[j] = 0.0f;
  return c;
}

// A-fragment (f16 16x32, row-major source [16][ld]):
// lane lo = row m; hi=0 -> K = kbase+0..7 and kbase+16..23; hi=1 -> +8 / +24.
__device__ __forceinline__ v16h load_a_frag(const half_t* base, int ld, int kbase,
                                            int lo, int hi) {
  const half_t* r = base + (size_t)lo * ld + kbase + hi * 8;
  v8h p0 = *(const v8h*)(r);
  v8h p1 = *(const v8h*)(r + 16);
  v16h a;
#pragma unroll
  for (int i = 0; i < 8; ++i) { a[i] = p0[i]; a[i + 8] = p1[i]; }
  return a;
}

// B-fragment (f16 32x16) from row-major weight [N_total][K_total] (i.e. W[out][in],
// computing X @ W^T): lane lo = output column n; hi=0 -> K 0..7 and 16..23; hi=1 -> +8 / +24.
__device__ __forceinline__ v16h load_b_frag(const half_t* w, int ld, int nbase,
                                            int kbase, int lo, int hi) {
  const half_t* r = w + (size_t)(nbase + lo) * ld + kbase + hi * 8;
  v8h p0 = *(const v8h*)(r);
  v8h p1 = *(const v8h*)(r + 16);
  v16h b;
#pragma unroll
  for (int i = 0; i < 8; ++i) { b[i] = p0[i]; b[i + 8] = p1[i]; }
  return b;
}

template <int KT>
__device__ __forceinline__ v8f gemm_tile(const half_t* abuf, int lda,
                                         const half_t* wmat, int ldw,
                                         int nbase, int lo, int hi) {
  v8f c = zero_v8f();
#pragma unroll
  for (int kt = 0; kt < KT; ++kt) {
    v16h a = load_a_frag(abuf, lda, kt * 32, lo, hi);
    v16h b = load_b_frag(wmat, ldw, nbase, kt * 32, lo, hi);
    c = __builtin_amdgcn_wmma_f32_16x16x32_f16(false, a, false, b, (short)0, c,
                                               false, false);
  }
  return c;
}

// C-fragment element (m, n): m = j + 8*hi, n = nbase + lo.
__device__ __forceinline__ void store_h_to_z(half_t* zbuf, const v8f hc[8],
                                             int lo, int hi) {
#pragma unroll
  for (int nt = 0; nt < 8; ++nt)
#pragma unroll
    for (int j = 0; j < 8; ++j)
      zbuf[(j + 8 * hi) * CIN + nt * 16 + lo] = (half_t)hc[nt][j];
}

// One ode_f eval, fused with RK4 bookkeeping:
//   k   = tanh(z @ W1^T + b1) @ W2^T + b2        (z in zbuf [16][192] f16)
//   acc += wacc * k                              (per C-fragment tile)
//   if WRITE_HE: zbuf_h = h + cf * dt * k        (input of the next eval)
// The k tile is consumed immediately; no kf[8] array stays live.
template <bool WRITE_HE>
__device__ __forceinline__ void ode_eval_fused(
    half_t* zbuf, half_t* tbuf, const half_t* sW1, const half_t* sW2,
    const float* sb1, const float* sb2, int lo, int hi,
    const v8f hc[8], v8f acc[8], const float dtv[8], float wacc, float cf) {
  // layer 1: u = tanh(z @ W1^T + b1) -> tbuf
#pragma unroll
  for (int nt = 0; nt < 8; ++nt) {
    v8f c = gemm_tile<6>(zbuf, CIN, sW1, CIN, nt * 16, lo, hi);
    float bv = sb1[nt * 16 + lo];
#pragma unroll
    for (int j = 0; j < 8; ++j)
      tbuf[(j + 8 * hi) * OHDIM + nt * 16 + lo] = (half_t)tanhf(c[j] + bv);
  }
  // layer 2: k tile -> acc (+ optional next h_e into zbuf cols 0..127)
#pragma unroll
  for (int nt = 0; nt < 8; ++nt) {
    v8f c = gemm_tile<4>(tbuf, OHDIM, sW2, OHDIM, nt * 16, lo, hi);
    float bv = sb2[nt * 16 + lo];
#pragma unroll
    for (int j = 0; j < 8; ++j) {
      float k = c[j] + bv;
      acc[nt][j] += wacc * k;
      if (WRITE_HE)
        zbuf[(j + 8 * hi) * CIN + nt * 16 + lo] =
            (half_t)(hc[nt][j] + cf * dtv[j] * k);
    }
  }
}

// Stage a [16][64] f32 tile of x (one timestep) into LDS as f16 (one wave).
__device__ __forceinline__ void stage_x_tile(const float* __restrict__ xg,
                                             int rowbase, int w, half_t* dst,
                                             int ldd, int coff, int lane) {
#pragma unroll
  for (int t = 0; t < 8; ++t) {
    int q = lane + t * 32;          // 256 float4 chunks
    int m = q >> 4;                 // row 0..15
    int k4 = (q & 15) * 4;          // col 0..60
    const float4 v =
        *(const float4*)(xg + ((size_t)(rowbase + m) * WLEN + w) * DXDIM + k4);
    v4h h4;
    h4[0] = (half_t)v.x; h4[1] = (half_t)v.y;
    h4[2] = (half_t)v.z; h4[3] = (half_t)v.w;
    *(v4h*)(dst + (size_t)m * ldd + coff + k4) = h4;
  }
}

__global__ void cvt_f32_to_f16(const float* __restrict__ src,
                               half_t* __restrict__ dst, int n) {
  int i = blockIdx.x * blockDim.x + threadIdx.x;
  if (i < n) dst[i] = (half_t)src[i];
}

__global__ __launch_bounds__(THREADS, 1) void odernn_scan_kernel(
    const float* __restrict__ x, const float* __restrict__ times,
    const float* __restrict__ Wih, const float* __restrict__ bih,
    const float* __restrict__ bhh, const float* __restrict__ W1,
    const float* __restrict__ b1, const float* __restrict__ W2,
    const float* __restrict__ b2, const float* __restrict__ resgate,
    const half_t* __restrict__ Whh16, float* __restrict__ out) {
  // LDS budget: 49152 + 32768 + 49152 + 4096 + 12288 + 8192 + 4096 + 256
  //           = 160,000 bytes (fits a 160KB CU-mode workgroup cap).
  __shared__ __align__(16) half_t sW1[OHDIM * CIN];
  __shared__ __align__(16) half_t sW2[HDIM * OHDIM];
  __shared__ __align__(16) half_t sWih[G3H * DXDIM];
  __shared__ float sb1[OHDIM], sb2[HDIM], sbih[G3H], sbhh[G3H];
  __shared__ __align__(16) half_t szbuf[WAVES_PER_BLOCK][16 * CIN];
  __shared__ __align__(16) half_t stbuf[WAVES_PER_BLOCK][16 * OHDIM];
  __shared__ __align__(16) half_t sxbuf[WAVES_PER_BLOCK][16 * DXDIM];
  __shared__ float sdt[WAVES_PER_BLOCK][16];
  __shared__ float sact[WAVES_PER_BLOCK][16];

  const int tid = threadIdx.x;
  const int wave = tid >> 5;
  const int lane = tid & 31;
  const int lo = lane & 15;
  const int hi = lane >> 4;

  for (int i = tid; i < OHDIM * CIN; i += THREADS) sW1[i] = (half_t)W1[i];
  for (int i = tid; i < HDIM * OHDIM; i += THREADS) sW2[i] = (half_t)W2[i];
  for (int i = tid; i < G3H * DXDIM; i += THREADS) sWih[i] = (half_t)Wih[i];
  for (int i = tid; i < OHDIM; i += THREADS) sb1[i] = b1[i];
  for (int i = tid; i < HDIM; i += THREADS) sb2[i] = b2[i];
  for (int i = tid; i < G3H; i += THREADS) { sbih[i] = bih[i]; sbhh[i] = bhh[i]; }
  __syncthreads();  // only workgroup barrier in the whole kernel

  const int rowbase = blockIdx.x * ROWS_PER_BLOCK + wave * 16;
  half_t* zbuf = szbuf[wave];   // [16][192]: cols 0..127 = h_e, 128..191 = ctrl
  half_t* tbuf = stbuf[wave];   // [16][128]: tanh(layer1) activations
  half_t* xbuf = sxbuf[wave];   // [16][64]:  x_t for the GRU
  float* dtb = sdt[wave];
  float* actb = sact[wave];

  const float alpha = 1.0f / (1.0f + __expf(-resgate[0]));
  const float beta = 1.0f - alpha;

  v8f hc[8];  // hidden state, 8 f32 C-fragments (16 rows x 128 cols)
#pragma unroll
  for (int nt = 0; nt < 8; ++nt) hc[nt] = zero_v8f();

#pragma unroll 1
  for (int w = 0; w < WLEN; ++w) {
    stage_x_tile(x, rowbase, w, xbuf, DXDIM, 0, lane);

    if (w > 0) {
      // ctrl = x_{t-1} into z-buffer tail (constant across the 8 evals)
      stage_x_tile(x, rowbase, w - 1, zbuf, CIN, HDIM, lane);

      // per-row dt_sub / substep-2 mask (row m lives at j + 8*hi in C frags)
      if (lane < 16) {
        const size_t tb = (size_t)(rowbase + lane) * WLEN;
        float dt = fmaxf(times[tb + w] - times[tb + w - 1], 0.0f);
        dt = fmaxf(dt, 1e-6f);
        float nf = fminf(fmaxf(ceilf(dt * 20.0f), 1.0f), 1000.0f);
        dtb[lane] = dt / nf;
        actb[lane] = (nf > 1.0f) ? 1.0f : 0.0f;
      }
      float dtv[8], av[8];
#pragma unroll
      for (int j = 0; j < 8; ++j) { dtv[j] = dtb[8 * hi + j]; av[j] = actb[8 * hi + j]; }

#pragma unroll 1
      for (int s = 0; s < 2; ++s) {  // RK4 substeps (row-masked on s==1)
        store_h_to_z(zbuf, hc, lo, hi);
        v8f acc[8];
#pragma unroll
        for (int nt = 0; nt < 8; ++nt) acc[nt] = zero_v8f();

        // k1..k4, each consumed in-place by the layer-2 tile loop
        ode_eval_fused<true >(zbuf, tbuf, sW1, sW2, sb1, sb2, lo, hi,
                              hc, acc, dtv, 1.0f, 0.5f);   // k1
        ode_eval_fused<true >(zbuf, tbuf, sW1, sW2, sb1, sb2, lo, hi,
                              hc, acc, dtv, 2.0f, 0.5f);   // k2
        ode_eval_fused<true >(zbuf, tbuf, sW1, sW2, sb1, sb2, lo, hi,
                              hc, acc, dtv, 2.0f, 1.0f);   // k3
        ode_eval_fused<false>(zbuf, tbuf, sW1, sW2, sb1, sb2, lo, hi,
                              hc, acc, dtv, 1.0f, 0.0f);   // k4

#pragma unroll
        for (int nt = 0; nt < 8; ++nt)
#pragma unroll
          for (int j = 0; j < 8; ++j) {
            float hn = hc[nt][j] + dtv[j] * (1.0f / 6.0f) * acc[nt][j];
            float gate = (s == 0) ? 1.0f : av[j];
            hc[nt][j] = (gate > 0.5f) ? hn : hc[nt][j];
          }
      }
    }

    // ---- GRU cell: h = (1-z)*n + z*h_ode, blended by alpha ----
    store_h_to_z(zbuf, hc, lo, hi);  // h_ode as A-operand (cols 0..127)
#pragma unroll
    for (int tg = 0; tg < 8; ++tg) {
      const int nr = tg * 16, nz = HDIM + tg * 16, nn = 2 * HDIM + tg * 16;
      v8f ir = gemm_tile<2>(xbuf, DXDIM, sWih, DXDIM, nr, lo, hi);
      v8f hr = gemm_tile<4>(zbuf, CIN, Whh16, HDIM, nr, lo, hi);
      v8f iz = gemm_tile<2>(xbuf, DXDIM, sWih, DXDIM, nz, lo, hi);
      v8f hz = gemm_tile<4>(zbuf, CIN, Whh16, HDIM, nz, lo, hi);
      v8f in_ = gemm_tile<2>(xbuf, DXDIM, sWih, DXDIM, nn, lo, hi);
      v8f hn = gemm_tile<4>(zbuf, CIN, Whh16, HDIM, nn, lo, hi);
      const float bir = sbih[nr + lo], bhr = sbhh[nr + lo];
      const float biz = sbih[nz + lo], bhz = sbhh[nz + lo];
      const float bin_ = sbih[nn + lo], bhn = sbhh[nn + lo];
#pragma unroll
      for (int j = 0; j < 8; ++j) {
        float r = 1.0f / (1.0f + __expf(-(ir[j] + bir + hr[j] + bhr)));
        float zg = 1.0f / (1.0f + __expf(-(iz[j] + biz + hz[j] + bhz)));
        float ng = tanhf(in_[j] + bin_ + r * (hn[j] + bhn));
        float hg = (1.0f - zg) * ng + zg * hc[tg][j];
        hc[tg][j] = alpha * hg + beta * hc[tg][j];
      }
    }
  }

  // write final hidden state [B, H] f32
#pragma unroll
  for (int nt = 0; nt < 8; ++nt)
#pragma unroll
    for (int j = 0; j < 8; ++j)
      out[(size_t)(rowbase + j + 8 * hi) * HDIM + nt * 16 + lo] = hc[nt][j];
}

extern "C" void kernel_launch(void* const* d_in, const int* in_sizes, int n_in,
                              void* d_out, int out_size, void* d_ws, size_t ws_size,
                              hipStream_t stream) {
  (void)in_sizes; (void)n_in; (void)out_size; (void)ws_size;
  const float* x     = (const float*)d_in[0];
  const float* times = (const float*)d_in[1];
  const float* Wih   = (const float*)d_in[2];
  const float* Whh   = (const float*)d_in[3];
  const float* bih   = (const float*)d_in[4];
  const float* bhh   = (const float*)d_in[5];
  const float* W1    = (const float*)d_in[6];
  const float* b1    = (const float*)d_in[7];
  const float* W2    = (const float*)d_in[8];
  const float* b2    = (const float*)d_in[9];
  const float* rgate = (const float*)d_in[10];
  float* out = (float*)d_out;

  // W_hh (384x128) -> f16 in workspace; stays L2-resident (96KB).
  half_t* whh16 = (half_t*)d_ws;
  const int nwhh = G3H * HDIM;
  cvt_f32_to_f16<<<(nwhh + 255) / 256, 256, 0, stream>>>(Whh, whh16, nwhh);

  odernn_scan_kernel<<<BATCH / ROWS_PER_BLOCK, THREADS, 0, stream>>>(
      x, times, Wih, bih, bhh, W1, b1, W2, b2, rgate, whh16, out);
}
